// CapsuleLayer_81879256531086
// MI455X (gfx1250) — compile-verified, matrix operand order checked by reading, not run on hardware
//
#include <hip/hip_runtime.h>

typedef __attribute__((ext_vector_type(16))) _Float16 v16h;
typedef __attribute__((ext_vector_type(8)))  float    v8f;

#define INC   32
#define ICS   8
#define ONC   32
#define OCS   16
#define KK    9
#define KKI   288      // KK*INC
#define BATCH 4
#define IH    14
#define IW    14
#define OH    12
#define OW    12
#define NTOT  576      // BATCH*OH*OW
#define NB    16       // n-rows per workgroup (WMMA M)
#define ITERS 3
#define NTHREADS 512   // 16 wave32 -> 4 waves per SIMD32 in the single resident WG

__global__ __launch_bounds__(NTHREADS, 1)
void caps_routing_kernel(const float* __restrict__ x,
                         const float* __restrict__ weight,
                         float* __restrict__ out) {
  // Dynamic LDS partition (floats):
  //   priors [KKI][NB][OCS]  = 73728
  //   logits [NB][KKI]       =  4608
  //   outv   [NB][OCS]       =   256
  //   ms     [NB][2]         =    32
  extern __shared__ float smem[];
  float* priors = smem;
  float* logits = smem + KKI * NB * OCS;
  float* outv   = logits + NB * KKI;
  float* ms     = outv + NB * OCS;

  const int tid  = threadIdx.x;
  const int lane = tid & 31;
  const int wave = tid >> 5;     // 0..15
  const int nb   = blockIdx.x;   // 0..35  (n-block)
  const int o    = blockIdx.y;   // 0..31  (output capsule)

  // Weight slab for this o is reused 36x across n-blocks: hint it into cache.
  __builtin_prefetch(weight + (size_t)o * KKI * ICS * OCS, 0, 3);

  // zero-init logits (barrier below covers this)
  for (int idx = tid; idx < NB * KKI; idx += NTHREADS) logits[idx] = 0.0f;

  // ---------------- Phase 1: priors via v_wmma_f32_16x16x32_f16 ----------------
  const int r  = lane & 15;       // M row (n) for A, column c for B
  const int hi = lane >> 4;

  // geometry of this lane's n-row (used by A loads, lanes 0-15)
  const int ng  = nb * NB + r;
  const int bi  = ng / (OH * OW);
  const int rem = ng % (OH * OW);
  const int ohh = rem / OW;
  const int oww = rem % OW;

  // Route mapping: inc = wave + 16*h fixed per pass, kk = 0..8 inner.
  // 16 waves x 2 passes x 9 kk = all 288 routes, 18 WMMAs per wave.
  #pragma unroll
  for (int h = 0; h < 2; ++h) {
    const int inc = wave + 16 * h;
    // per-(lane, inc) x base: channel block inc*ICS, spatial (ohh, oww)
    const float* xbase =
        x + ((size_t)(bi * (INC * ICS) + inc * ICS) * IH + ohh) * IW + oww;
    const float* wbase =
        weight + ((size_t)(o * KKI + inc) * ICS) * OCS + r;

    for (int kk = 0; kk < KK; ++kk) {
      const int kh = kk / 3, kw = kk % 3;      // cheap: 9 outer iterations
      const int k  = kk * INC + inc;

      // A: 16x32 f16, lanes0-15 elems0..7 = K=0..7 (= i), everything else 0.
      // B: 32x16 f16, lanes0-15 elems0..7 = K rows 0..7, N = lane; rest 0.
      v16h a = {};
      v16h b = {};
      if (hi == 0) {
        const float* xp = xbase + kh * IW + kw;
        const float* wp = wbase + (size_t)kk * (INC * ICS * OCS);
        #pragma unroll
        for (int i = 0; i < ICS; ++i) {
          a[i] = (_Float16)xp[i * (IH * IW)];
          b[i] = (_Float16)wp[i * OCS];
        }
      }
      v8f cz = {};
      v8f d = __builtin_amdgcn_wmma_f32_16x16x32_f16(false, a, false, b,
                                                     (short)0, cz, false, false);
      // D layout: elem j -> (n = j + 8*hi, c = lane&15)
      float* pk = priors + ((size_t)k * NB) * OCS + (lane & 15);
      #pragma unroll
      for (int j = 0; j < 8; ++j)
        pk[(j + 8 * hi) * OCS] = d[j];
    }
  }
  __syncthreads();   // priors (and logits init) visible to all waves

  // ---------------- Phase 2: dynamic routing (3 iterations) ----------------
  // 16 waves, wave w exclusively owns n-row nn = w -> no further barriers.
  const int nn = wave;
  for (int iter = 0; iter < ITERS; ++iter) {
    // softmax stats over the 288 routes; lane = k mod 32
    {
      float lv[KKI / 32];
      float m = -3.0e38f;
      #pragma unroll
      for (int j = 0; j < KKI / 32; ++j) {
        lv[j] = logits[nn * KKI + j * 32 + lane];
        m = fmaxf(m, lv[j]);
      }
      #pragma unroll
      for (int off = 16; off >= 1; off >>= 1)
        m = fmaxf(m, __shfl_xor(m, off, 32));
      float s = 0.0f;
      #pragma unroll
      for (int j = 0; j < KKI / 32; ++j) s += __expf(lv[j] - m);
      #pragma unroll
      for (int off = 16; off >= 1; off >>= 1)
        s += __shfl_xor(s, off, 32);
      if (lane == 0) { ms[nn * 2] = m; ms[nn * 2 + 1] = 1.0f / s; }
    }

    // weighted sum + squash; lanes = c(16) x k-half(2)
    {
      const int c = lane & 15, kh2 = lane >> 4;
      const float m  = ms[nn * 2];
      const float is = ms[nn * 2 + 1];
      float acc = 0.0f;
      for (int j = 0; j < KKI / 2; ++j) {
        const int k = kh2 * (KKI / 2) + j;
        const float p = __expf(logits[nn * KKI + k] - m) * is;
        acc = fmaf(p, priors[((size_t)k * NB + nn) * OCS + c], acc);
      }
      acc += __shfl_xor(acc, 16, 32);              // combine k-halves
      float sq = acc * acc;                         // sum over c (16 lanes)
      #pragma unroll
      for (int off = 1; off <= 8; off <<= 1)
        sq += __shfl_xor(sq, off, 32);
      const float scale = (sq / (1.0f + sq)) * __frsqrt_rn(sq);
      const float oc = acc * scale;
      if (kh2 == 0) {
        if (iter == ITERS - 1) {
          const int ng2 = nb * NB + nn;
          const int b2 = ng2 / (OH * OW), r2 = ng2 % (OH * OW);
          out[((size_t)(b2 * (ONC * OCS) + o * OCS + c) * OH + r2 / OW) * OW
              + (r2 % OW)] = oc;
        } else {
          outv[nn * OCS + c] = oc;
        }
      }
    }

    // logit update; lane = k mod 32
    if (iter != ITERS - 1) {
      #pragma unroll
      for (int j = 0; j < KKI / 32; ++j) {
        const int k = j * 32 + lane;
        float dot = 0.0f;
        #pragma unroll
        for (int c = 0; c < OCS; ++c)
          dot = fmaf(priors[((size_t)k * NB + nn) * OCS + c],
                     outv[nn * OCS + c], dot);
        logits[nn * KKI + k] += dot;
      }
    }
  }
}

extern "C" void kernel_launch(void* const* d_in, const int* in_sizes, int n_in,
                              void* d_out, int out_size, void* d_ws, size_t ws_size,
                              hipStream_t stream) {
  const float* x = (const float*)d_in[0];      // (4, 256, 14, 14)
  const float* w = (const float*)d_in[1];      // (32, 288, 8, 16)
  float* out = (float*)d_out;                  // (4, 512, 12, 12)

  const size_t shmem =
      (size_t)(KKI * NB * OCS + NB * KKI + NB * OCS + 32) * sizeof(float); // 314496 B
  hipFuncSetAttribute(reinterpret_cast<const void*>(caps_routing_kernel),
                      hipFuncAttributeMaxDynamicSharedMemorySize, (int)shmem);

  dim3 grid(NTOT / NB, ONC);     // 36 x 32 workgroups
  dim3 block(NTHREADS);          // 16 wave32
  caps_routing_kernel<<<grid, block, shmem, stream>>>(x, w, out);
}
